// CEHessianCalculator_46213848105703
// MI455X (gfx1250) — matrix operand order repaired and checked by reading, not compiled, not run on hardware
//
#include <hip/hip_runtime.h>

// ---------------------------------------------------------------------------
// CE Hessian (diagonal, wrt input) for MI455X / gfx1250 (wave32, WMMA).
//   out[d] = mean_b( (e@W^2)[b,d]/S[b] - ((e@W)[b,d]/S[b])^2 ),  e = exp(x@W^T + b)
// Single pass over C; bf16 WMMA (v_wmma_f32_16x16x32_bf16) with f32 accumulation.
// Fast path: W pre-converted once to bf16 (row-major + transposed + transposed^2)
// in d_ws; B-fragments stream straight from L2 (all bf16 W arrays ~38.7MB < 192MB L2).
// Fallback path (small ws): fully fused kernel with in-LDS conversion.
// ---------------------------------------------------------------------------

typedef __attribute__((ext_vector_type(16))) __bf16      v16bf;
typedef __attribute__((ext_vector_type(8)))  float        v8f;
typedef __attribute__((ext_vector_type(4)))  unsigned int v4u;
typedef unsigned short ushort_t;

#define B_ROWS 4096
#define C_DIM  50257
#define D_DIM  128
#define CT     128                       // C-chunk per iteration
#define NCH    ((C_DIM + CT - 1) / CT)   // 393 chunks
#define CPAD   (NCH * CT)                // 50304 zero-padded rows
#define ST     136                       // LDS row stride (bf16 elems, 16B aligned)

#define BTF    32                        // fast path: B rows per WG
#define NWGF   (B_ROWS / BTF)            // 128 workgroups
#define BTS    16                        // fallback: B rows per WG
#define NWGS   (B_ROWS / BTS)            // 256 workgroups

__device__ __forceinline__ ushort_t f2bf(float f) {
  unsigned int u = __builtin_bit_cast(unsigned int, f);
  u += 0x7FFFu + ((u >> 16) & 1u);       // round-to-nearest-even
  return (ushort_t)(u >> 16);
}

// 32-byte bf16 fragment from two 16-byte loads (ds_load_b128 / global_load_b128).
__device__ __forceinline__ v16bf ld_frag(const ushort_t* p0, const ushort_t* p1) {
  v16bf r;
  v4u* pr = (v4u*)&r;
  pr[0] = *(const v4u*)p0;
  pr[1] = *(const v4u*)p1;
  return r;
}

__device__ __forceinline__ v8f wmma_bf16(v16bf a, v16bf bm, v8f c) {
  return __builtin_amdgcn_wmma_f32_16x16x32_bf16(false, a, false, bm, (short)0, c, false, false);
}

// ---------------------------------------------------------------------------
// Prep: W fp32 [C][128] -> Wbf bf16 [CPAD][128] (row major, zero padded)
//                       -> Wtg bf16 [128][CPAD] (transposed)
//                       -> Wt2g bf16 [128][CPAD] (transposed, squared)
// One 128x128 tile per workgroup; transpose staged through LDS.
// ---------------------------------------------------------------------------
__global__ void __launch_bounds__(256)
ce_prep_w(const float* __restrict__ W, ushort_t* __restrict__ Wbf,
          ushort_t* __restrict__ Wtg, ushort_t* __restrict__ Wt2g)
{
  __shared__ ushort_t t1[128 * 132];     // transposed bf16 tile [d][c], skewed stride
  const int tid = threadIdx.x;
  const int cb  = blockIdx.x * 128;
  {
    const int r = tid >> 1, h = tid & 1; // r = local c row, h = 64-wide D half
    const int c = cb + r;
    const bool ok = (c < C_DIM);
    const float4* wp = ok ? (const float4*)(W + (size_t)c * D_DIM + h * 64) : (const float4*)0;
    unsigned int* qo = (unsigned int*)(Wbf + (size_t)c * D_DIM + h * 64);
    for (int j = 0; j < 16; ++j) {
      float4 f = ok ? wp[j] : make_float4(0.f, 0.f, 0.f, 0.f);
      const int d = h * 64 + j * 4;
      const ushort_t b0 = f2bf(f.x), b1 = f2bf(f.y), b2 = f2bf(f.z), b3 = f2bf(f.w);
      qo[2 * j]     = (unsigned int)b0 | ((unsigned int)b1 << 16);
      qo[2 * j + 1] = (unsigned int)b2 | ((unsigned int)b3 << 16);
      t1[(d + 0) * 132 + r] = b0;
      t1[(d + 1) * 132 + r] = b1;
      t1[(d + 2) * 132 + r] = b2;
      t1[(d + 3) * 132 + r] = b3;
    }
  }
  __syncthreads();
  {
    const int dr = tid >> 1, h = tid & 1;
    const ushort_t* src = &t1[dr * 132 + h * 64];
    ushort_t* o1 = Wtg  + (size_t)dr * CPAD + cb + h * 64;
    ushort_t* o2 = Wt2g + (size_t)dr * CPAD + cb + h * 64;
    for (int j = 0; j < 64; j += 2) {
      const ushort_t v0 = src[j], v1 = src[j + 1];
      *(unsigned int*)(o1 + j) = (unsigned int)v0 | ((unsigned int)v1 << 16);
      const float f0 = __builtin_bit_cast(float, (unsigned int)v0 << 16);
      const float f1 = __builtin_bit_cast(float, (unsigned int)v1 << 16);
      *(unsigned int*)(o2 + j) =
          (unsigned int)f2bf(f0 * f0) | ((unsigned int)f2bf(f1 * f1) << 16);
    }
  }
}

// ---------------------------------------------------------------------------
// Fast main: 32-row B-tile per WG, streams C in 128 chunks; 24 WMMAs/wave/chunk.
// B-operands load straight from L2-resident bf16 arrays; LDS holds only x & P.
// ---------------------------------------------------------------------------
__global__ void __launch_bounds__(256)
ce_main_fast(const float* __restrict__ x, const float* __restrict__ bias,
             const ushort_t* __restrict__ Wbf, const ushort_t* __restrict__ Wtg,
             const ushort_t* __restrict__ Wt2g, float* __restrict__ wsp)
{
  __shared__ ushort_t xlds[BTF * ST];
  __shared__ ushort_t Plds[BTF * ST];
  __shared__ float    Srow[BTF];

  const int tid  = threadIdx.x;
  const int lane = tid & 31;
  const int wv   = tid >> 5;     // wave 0..7 : GEMM1 c-tile, GEMM2 d-tile
  const int n    = lane & 15;    // tile column / A-matrix row
  const int kp   = lane >> 4;    // half-wave K selector
  const int wg   = blockIdx.x;
  const int row0 = wg * BTF;

  {  // stage x tile: 32x128 fp32 -> bf16 LDS (16 elems per thread)
    const int idx = tid * 16;
    const int r = idx >> 7, d = idx & 127;
    const float* xp = x + (size_t)(row0 + r) * D_DIM + d;
    for (int j = 0; j < 16; ++j) xlds[r * ST + d + j] = f2bf(xp[j]);
  }
  if (tid < BTF) Srow[tid] = 0.0f;
  __syncthreads();

  // x A-fragments for both M-tiles, kept in VGPRs for the whole C loop.
  v16bf xf[2][4];
  for (int mt = 0; mt < 2; ++mt)
    for (int k = 0; k < 4; ++k) {
      const ushort_t* p = &xlds[(mt * 16 + n) * ST + k * 32 + kp * 8];
      xf[mt][k] = ld_frag(p, p + 16);
    }

  v8f accA[2], accA2[2];
  for (int mt = 0; mt < 2; ++mt)
    for (int i = 0; i < 8; ++i) { accA[mt][i] = 0.0f; accA2[mt][i] = 0.0f; }

  for (int ch = 0; ch < NCH; ++ch) {
    const int c0 = ch * CT;
    __syncthreads();                       // previous chunk's Plds readers done

    // ---- GEMM1: logits for this wave's 16 c-columns, both M-tiles ----
    const int cg = c0 + wv * 16 + n;                 // < CPAD always
    const float bv = (cg < C_DIM) ? bias[cg] : -1e30f;  // pad cols -> exp()=0
    const ushort_t* wrow = Wbf + (size_t)cg * D_DIM;    // contiguous K=d run
    for (int mt = 0; mt < 2; ++mt) {
      v8f lg;
      for (int i = 0; i < 8; ++i) lg[i] = bv;
      for (int k = 0; k < 4; ++k) {
        const ushort_t* p = wrow + k * 32 + kp * 8;
        lg = wmma_bf16(xf[mt][k], ld_frag(p, p + 16), lg);
      }
      v8f e;
      for (int i = 0; i < 8; ++i)
        e[i] = __builtin_amdgcn_exp2f(lg[i] * 1.44269504088896f);
      for (int i = 0; i < 8; ++i)
        Plds[(mt * 16 + i + 8 * kp) * ST + wv * 16 + n] = f2bf(e[i]);
      for (int i = 0; i < 8; ++i) {        // row sums across the 16 N-lanes
        float s = e[i];
        s += __shfl_xor(s, 1, 32);
        s += __shfl_xor(s, 2, 32);
        s += __shfl_xor(s, 4, 32);
        s += __shfl_xor(s, 8, 32);
        if (n == 0) atomicAdd(&Srow[mt * 16 + i + 8 * kp], s);
      }
    }
    __syncthreads();                       // Plds complete

    // ---- GEMM2: accA += P@W, accA2 += P@W^2 for this wave's 16 d-columns ----
    const size_t wtrow = (size_t)(wv * 16 + n) * CPAD + c0;
    for (int k = 0; k < 4; ++k) {
      const ushort_t* bp  = Wtg  + wtrow + k * 32 + kp * 8;
      const ushort_t* bp2 = Wt2g + wtrow + k * 32 + kp * 8;
      const v16bf b1 = ld_frag(bp,  bp  + 16);   // shared by both M-tiles
      const v16bf b2 = ld_frag(bp2, bp2 + 16);
      for (int mt = 0; mt < 2; ++mt) {
        const ushort_t* ap = &Plds[(mt * 16 + n) * ST + k * 32 + kp * 8];
        const v16bf af = ld_frag(ap, ap + 16);
        accA[mt]  = wmma_bf16(af, b1, accA[mt]);
        accA2[mt] = wmma_bf16(af, b2, accA2[mt]);
      }
    }
    if (ch + 1 < NCH) {                    // gfx1250 global_prefetch_b8
      __builtin_prefetch(Wtg  + wtrow + CT, 0, 1);
      __builtin_prefetch(Wt2g + wtrow + CT, 0, 1);
    }
  }

  __syncthreads();
  float colsum = 0.0f;
  for (int mt = 0; mt < 2; ++mt)
    for (int i = 0; i < 8; ++i) {          // element (m = mt*16+i+8*kp, d = wv*16+n)
      const float S = Srow[mt * 16 + i + 8 * kp];
      const float a = accA[mt][i] / S;
      colsum += accA2[mt][i] / S - a * a;
    }
  colsum += __shfl_xor(colsum, 16, 32);    // fold other half-wave's rows
  if (lane < 16)
    wsp[(size_t)wg * D_DIM + wv * 16 + lane] = colsum;
}

// ---------------------------------------------------------------------------
// Fallback fused kernel (small d_ws): in-LDS W conversion, 16-row B-tile.
// ---------------------------------------------------------------------------
__global__ void __launch_bounds__(256)
ce_main_fused(const float* __restrict__ x, const float* __restrict__ W,
              const float* __restrict__ bias, float* __restrict__ wsp)
{
  extern __shared__ ushort_t smem[];
  ushort_t* xlds = smem;                   // [BTS][ST]
  ushort_t* Plds = xlds + BTS * ST;        // [BTS][ST]
  ushort_t* Wrow = Plds + BTS * ST;        // [CT][ST]  c-major
  ushort_t* Wt   = Wrow + CT * ST;         // [D ][ST]  d-major
  ushort_t* Wt2  = Wt   + D_DIM * ST;      // [D ][ST]  d-major squared
  float*    Srow = (float*)(Wt2 + D_DIM * ST);

  const int tid  = threadIdx.x;
  const int lane = tid & 31;
  const int wv   = tid >> 5;
  const int n    = lane & 15;
  const int kp   = lane >> 4;
  const int row0 = blockIdx.x * BTS;

  {
    const int idx = tid * 8;
    const int r = idx >> 7, d = idx & 127;
    const float* xp = x + (size_t)(row0 + r) * D_DIM + d;
    for (int j = 0; j < 8; ++j) xlds[r * ST + d + j] = f2bf(xp[j]);
  }
  if (tid < BTS) Srow[tid] = 0.0f;
  __syncthreads();

  v16bf xf[4];
  for (int k = 0; k < 4; ++k) {
    const ushort_t* p = &xlds[n * ST + k * 32 + kp * 8];
    xf[k] = ld_frag(p, p + 16);
  }
  v8f accA, accA2;
  for (int i = 0; i < 8; ++i) { accA[i] = 0.0f; accA2[i] = 0.0f; }

  for (int ch = 0; ch < NCH; ++ch) {
    const int c0 = ch * CT;
    __syncthreads();
    {
      const int r = tid >> 1, h = tid & 1;
      const int c = c0 + r;
      const bool ok = (c < C_DIM);
      const float4* wp = ok ? (const float4*)(W + (size_t)c * D_DIM + h * 64) : (const float4*)0;
      for (int j = 0; j < 16; ++j) {
        float4 f = ok ? wp[j] : make_float4(0.f, 0.f, 0.f, 0.f);
        const int d = h * 64 + j * 4;
        const ushort_t b0 = f2bf(f.x), b1 = f2bf(f.y), b2 = f2bf(f.z), b3 = f2bf(f.w);
        unsigned int* q = (unsigned int*)&Wrow[r * ST + d];
        q[0] = (unsigned int)b0 | ((unsigned int)b1 << 16);
        q[1] = (unsigned int)b2 | ((unsigned int)b3 << 16);
        Wt [(d + 0) * ST + r] = b0;  Wt [(d + 1) * ST + r] = b1;
        Wt [(d + 2) * ST + r] = b2;  Wt [(d + 3) * ST + r] = b3;
        Wt2[(d + 0) * ST + r] = f2bf(f.x * f.x);
        Wt2[(d + 1) * ST + r] = f2bf(f.y * f.y);
        Wt2[(d + 2) * ST + r] = f2bf(f.z * f.z);
        Wt2[(d + 3) * ST + r] = f2bf(f.w * f.w);
      }
    }
    __syncthreads();

    const int cg = c0 + wv * 16 + n;
    const float bv = (cg < C_DIM) ? bias[cg] : -1e30f;
    v8f lg;
    for (int i = 0; i < 8; ++i) lg[i] = bv;
    for (int k = 0; k < 4; ++k) {
      const ushort_t* p = &Wrow[(wv * 16 + n) * ST + k * 32 + kp * 8];
      lg = wmma_bf16(xf[k], ld_frag(p, p + 16), lg);
    }
    v8f e;
    for (int i = 0; i < 8; ++i) e[i] = __builtin_amdgcn_exp2f(lg[i] * 1.44269504088896f);
    for (int i = 0; i < 8; ++i) Plds[(i + 8 * kp) * ST + wv * 16 + n] = f2bf(e[i]);
    for (int i = 0; i < 8; ++i) {
      float s = e[i];
      s += __shfl_xor(s, 1, 32); s += __shfl_xor(s, 2, 32);
      s += __shfl_xor(s, 4, 32); s += __shfl_xor(s, 8, 32);
      if (n == 0) atomicAdd(&Srow[i + 8 * kp], s);
    }
    __syncthreads();

    for (int k = 0; k < 4; ++k) {
      const ushort_t* ap  = &Plds[n * ST + k * 32 + kp * 8];
      const v16bf af = ld_frag(ap, ap + 16);
      const ushort_t* bp  = &Wt [(wv * 16 + n) * ST + k * 32 + kp * 8];
      const ushort_t* bp2 = &Wt2[(wv * 16 + n) * ST + k * 32 + kp * 8];
      accA  = wmma_bf16(af, ld_frag(bp,  bp  + 16), accA);
      accA2 = wmma_bf16(af, ld_frag(bp2, bp2 + 16), accA2);
    }
  }

  __syncthreads();
  float colsum = 0.0f;
  for (int i = 0; i < 8; ++i) {
    const float S = Srow[i + 8 * kp];
    const float a = accA[i] / S;
    colsum += accA2[i] / S - a * a;
  }
  colsum += __shfl_xor(colsum, 16, 32);
  if (lane < 16)
    wsp[(size_t)blockIdx.x * D_DIM + wv * 16 + lane] = colsum;
}

__global__ void __launch_bounds__(128)
ce_reduce(const float* __restrict__ wsp, float* __restrict__ out, int nwg)
{
  const int d = threadIdx.x;
  float s = 0.0f;
  for (int w = 0; w < nwg; ++w) s += wsp[(size_t)w * D_DIM + d];
  out[d] = s * (1.0f / (float)B_ROWS);
}

extern "C" void kernel_launch(void* const* d_in, const int* in_sizes, int n_in,
                              void* d_out, int out_size, void* d_ws, size_t ws_size,
                              hipStream_t stream) {
  (void)in_sizes; (void)n_in; (void)out_size;
  const float* x = (const float*)d_in[0];
  const float* W = (const float*)d_in[1];
  const float* b = (const float*)d_in[2];
  float* out = (float*)d_out;
  float* wsp = (float*)d_ws;

  const size_t PART   = (size_t)NWGS * D_DIM * sizeof(float);   // 128 KB (covers both paths)
  const size_t WBYTES = (size_t)CPAD * D_DIM * sizeof(ushort_t);
  const size_t NEEDED = PART + 3 * WBYTES;                      // ~38.9 MB

  if (ws_size >= NEEDED) {
    ushort_t* Wbf  = (ushort_t*)((char*)d_ws + PART);
    ushort_t* Wtg  = Wbf + (size_t)CPAD * D_DIM;
    ushort_t* Wt2g = Wtg + (size_t)CPAD * D_DIM;
    ce_prep_w<<<NCH, 256, 0, stream>>>(W, Wbf, Wtg, Wt2g);
    ce_main_fast<<<NWGF, 256, 0, stream>>>(x, b, Wbf, Wtg, Wt2g, wsp);
    ce_reduce<<<1, D_DIM, 0, stream>>>(wsp, out, NWGF);
  } else {
    const size_t shmem = (size_t)(2 * BTS * ST + 3 * CT * ST) * sizeof(ushort_t)
                       + BTS * sizeof(float);                   // 113,216 B
    hipFuncSetAttribute((const void*)ce_main_fused,
                        hipFuncAttributeMaxDynamicSharedMemorySize, (int)shmem);
    ce_main_fused<<<NWGS, 256, shmem, stream>>>(x, W, b, wsp);
    ce_reduce<<<1, D_DIM, 0, stream>>>(wsp, out, NWGS);
  }
}
